// TransformerEncoderLayer_10230612099074
// MI455X (gfx1250) — matrix operand order repaired and checked by reading, not compiled, not run on hardware
//
#include <hip/hip_runtime.h>
#include <hip/hip_bf16.h>

// ---------------- problem constants ----------------
#define S_LEN   2048
#define B_SZ    2
#define D_MODEL 1024
#define NH      16
#define HD      64
#define FF      4096
#define NROW    (S_LEN * B_SZ)   // 4096 rows, row-major [NROW][D]

typedef __attribute__((ext_vector_type(16))) __bf16 v16bf;
typedef __attribute__((ext_vector_type(8)))  __bf16 v8bf;
typedef __attribute__((ext_vector_type(8)))  float  v8f;

union V16 { v16bf v; v8bf h[2]; };

__device__ __forceinline__ unsigned short f2bf(float f) {
  unsigned int u = __float_as_uint(f);
  u = (u + 0x7FFFu + ((u >> 16) & 1u)) >> 16;   // round-to-nearest-even
  return (unsigned short)u;
}

__device__ __forceinline__ v8f wmma_bf16(v16bf a, v16bf b, v8f c) {
  return __builtin_amdgcn_wmma_f32_16x16x32_bf16(false, a, false, b, (short)0, c,
                                                 false, false);
}

// LDS byte offset of a generic pointer to __shared__ (flat-LDS addr low 32 bits
// are the DS offset per ISA 10.2 aperture mapping)
__device__ __forceinline__ unsigned lds_off(const void* p) {
  return (unsigned)(unsigned long long)p;
}

// CDNA5 async DMA: global -> LDS, 16 bytes per lane, tracked by ASYNCcnt
__device__ __forceinline__ void async_b128(unsigned lds, const void* g) {
  asm volatile("global_load_async_to_lds_b128 %0, %1, off"
               :: "v"(lds), "v"(g) : "memory");
}
__device__ __forceinline__ void wait_async0() {
  asm volatile("s_wait_asynccnt 0x0" ::: "memory");
}
__device__ __forceinline__ void wait_async4() {   // previous tile (4 ops) complete
  asm volatile("s_wait_asynccnt 0x4" ::: "memory");
}

// ---------------- converts ----------------
__global__ __launch_bounds__(256) void k_f32_to_bf16(const float* __restrict__ in,
                                                     unsigned short* __restrict__ out,
                                                     int n) {
  int i = blockIdx.x * blockDim.x + threadIdx.x;
  if (i < n) out[i] = f2bf(in[i]);
}

// W[K][N] fp32 -> Wt[N][K] bf16
__global__ __launch_bounds__(256) void k_transpose_to_bf16(const float* __restrict__ W,
                                                           unsigned short* __restrict__ Wt,
                                                           int K, int N) {
  int i = blockIdx.x * blockDim.x + threadIdx.x;
  if (i < K * N) {
    int k = i / N, n = i % N;
    Wt[(size_t)n * K + k] = f2bf(W[i]);
  }
}

// Vbb[(s*B+b)][h*HD+d] bf16 -> Vt[((b*NH+h)*HD + d)][s] bf16  ([2048][2048])
__global__ __launch_bounds__(256) void k_transpose_v(const unsigned short* __restrict__ V,
                                                     unsigned short* __restrict__ Vt) {
  int o = blockIdx.x * blockDim.x + threadIdx.x;   // over 2048*2048
  int s = o & (S_LEN - 1);
  int r = o >> 11;                 // (b*NH+h)*HD + d
  int d = r & (HD - 1);
  int bh = r >> 6;                 // b*NH + h
  int b  = bh >> 4;
  int h  = bh & (NH - 1);
  Vt[o] = V[((size_t)s * B_SZ + b) * D_MODEL + h * HD + d];
}

// ---------------- generic bf16 GEMM ----------------
// out = act((A[M][K] x Bt[N][K]^T + bias)*alpha) + resid ; async double-buffered tiles.
// Last K-step is peeled so the steady-state loop is branch-free.
__global__ __launch_bounds__(256) void k_gemm(const unsigned short* __restrict__ A,
                                              const unsigned short* __restrict__ Bt,
                                              const float* __restrict__ bias,
                                              float alpha, int relu,
                                              const float* __restrict__ resid,
                                              float* __restrict__ outF,
                                              unsigned short* __restrict__ outB,
                                              int M, int N, int K) {
  __shared__ unsigned short sA[2][128][32];
  __shared__ unsigned short sB[2][128][32];

  const int tid  = threadIdx.x;
  const int wave = tid >> 5;
  const int lane = tid & 31;
  const int laneHalf = lane >> 4;
  const int lane16   = lane & 15;
  const int m0 = blockIdx.y * 128;
  const int n0 = blockIdx.x * 128;
  const int wm = wave >> 1;   // 0..3 -> 32-row strip
  const int wn = wave & 1;    // 0..1 -> 64-col strip

  const int lr = tid >> 1;            // 0..127 tile row this thread loads
  const int lc = (tid & 1) * 16;      // 0/16 half-offset (16 halfs = 32B)

  const unsigned la[2] = { lds_off(&sA[0][lr][lc]), lds_off(&sA[1][lr][lc]) };
  const unsigned lb[2] = { lds_off(&sB[0][lr][lc]), lds_off(&sB[1][lr][lc]) };
  const unsigned short* gA = &A[(size_t)(m0 + lr) * K + lc];
  const unsigned short* gB = &Bt[(size_t)(n0 + lr) * K + lc];

  v8f zero = {0.f, 0.f, 0.f, 0.f, 0.f, 0.f, 0.f, 0.f};
  v8f acc[2][4];
  for (int mi = 0; mi < 2; ++mi)
    for (int ni = 0; ni < 4; ++ni) acc[mi][ni] = zero;

  const int akb = laneHalf * 8;
  const int bkb = laneHalf * 8;
  auto compute = [&](int buf) {
    V16 af[2];
    for (int mi = 0; mi < 2; ++mi) {
      int r = wm * 32 + mi * 16 + lane16;
      af[mi].h[0] = *reinterpret_cast<const v8bf*>(&sA[buf][r][akb]);
      af[mi].h[1] = *reinterpret_cast<const v8bf*>(&sA[buf][r][akb + 16]);
    }
    V16 bfr[4];
    for (int ni = 0; ni < 4; ++ni) {
      int c = wn * 64 + ni * 16 + lane16;
      bfr[ni].h[0] = *reinterpret_cast<const v8bf*>(&sB[buf][c][bkb]);
      bfr[ni].h[1] = *reinterpret_cast<const v8bf*>(&sB[buf][c][bkb + 16]);
    }
    for (int mi = 0; mi < 2; ++mi)
      for (int ni = 0; ni < 4; ++ni)
        acc[mi][ni] = wmma_bf16(af[mi].v, bfr[ni].v, acc[mi][ni]);
  };

  // prologue: tile 0 in flight
  async_b128(la[0],      gA);
  async_b128(la[0] + 16, gA + 8);
  async_b128(lb[0],      gB);
  async_b128(lb[0] + 16, gB + 8);

  const int NK = K >> 5;
  for (int i = 0; i < NK - 1; ++i) {          // steady state: branch-free
    const int buf = i & 1;
    const int nb  = buf ^ 1;
    const unsigned short* gA1 = gA + (size_t)(i + 1) * 32;
    const unsigned short* gB1 = gB + (size_t)(i + 1) * 32;
    async_b128(la[nb],      gA1);
    async_b128(la[nb] + 16, gA1 + 8);
    async_b128(lb[nb],      gB1);
    async_b128(lb[nb] + 16, gB1 + 8);
    wait_async4();                            // tile i landed (in-order)
    __syncthreads();
    compute(buf);
    __syncthreads();                          // buf free for refill next iter
  }
  wait_async0();                              // peeled last tile
  __syncthreads();
  compute((NK - 1) & 1);

  // epilogue; C layout: vgpr r, lanes<16 -> M=r, lanes>=16 -> M=8+r, N=lane16
  for (int mi = 0; mi < 2; ++mi) {
    int rowBase = m0 + wm * 32 + mi * 16 + laneHalf * 8;
    for (int ni = 0; ni < 4; ++ni) {
      int col = n0 + wn * 64 + ni * 16 + lane16;
      float bv = bias ? bias[col] : 0.0f;
      for (int r = 0; r < 8; ++r) {
        size_t idx = (size_t)(rowBase + r) * N + col;
        float v = (acc[mi][ni][r] + bv) * alpha;
        if (relu) v = fmaxf(v, 0.0f);
        if (resid) v += resid[idx];
        if (outF) outF[idx] = v;
        if (outB) outB[idx] = f2bf(v);
      }
    }
  }
}

// ---------------- flash attention ----------------
// grid: (B*H, S/128). block 256 = 8 waves; wave owns 16 query rows.
// K tiles from Kbb[(s*B+b)][...]; V tiles from pre-transposed Vt[(b,h,dim)][s].
__global__ __launch_bounds__(256) void k_attention(const unsigned short* __restrict__ Qb,
                                                   const unsigned short* __restrict__ Kb,
                                                   const unsigned short* __restrict__ Vt,
                                                   unsigned short* __restrict__ ctx) {
  __shared__ unsigned short sQ[128][64];      // [qrow][dim]          16KB
  __shared__ unsigned short sK[2][64][64];    // [key][dim]           16KB
  __shared__ unsigned short sVt[2][64][64];   // [dim][key]           16KB
  __shared__ unsigned short sP[128][64];      // P staging            16KB

  const int tid  = threadIdx.x;
  const int wave = tid >> 5;
  const int lane = tid & 31;
  const int laneHalf = lane >> 4;
  const int lane16   = lane & 15;

  const int bh = blockIdx.x;
  const int bb = bh % B_SZ;
  const int hh = bh / B_SZ;
  const int q0 = blockIdx.y * 128;
  const int colBase = hh * HD;
  const int vtRowBase = (bb * NH + hh) * HD;

  // ---- async prologue: Q tile (4 ops) then KV tile 0 (4 ops) ----
  {
    int qi = tid >> 1;
    int qc = (tid & 1) * 32;
    const unsigned short* gq = &Qb[((size_t)(q0 + qi) * B_SZ + bb) * D_MODEL + colBase + qc];
    unsigned lq = lds_off(&sQ[qi][qc]);
    async_b128(lq,      gq);
    async_b128(lq + 16, gq + 8);
    async_b128(lq + 32, gq + 16);
    async_b128(lq + 48, gq + 24);
  }
  const int kd = tid >> 2;          // 0..63 (key row for K, dim row for Vt)
  const int kc = (tid & 3) * 16;    // 16-half chunk
  const unsigned lk[2] = { lds_off(&sK[0][kd][kc]),  lds_off(&sK[1][kd][kc]) };
  const unsigned lv[2] = { lds_off(&sVt[0][kd][kc]), lds_off(&sVt[1][kd][kc]) };
  const unsigned short* gV = &Vt[(size_t)(vtRowBase + kd) * S_LEN + kc];

  auto issueKV = [&](int kb0, int buf) {
    const unsigned short* gk = &Kb[((size_t)(kb0 + kd) * B_SZ + bb) * D_MODEL + colBase + kc];
    async_b128(lk[buf],      gk);
    async_b128(lk[buf] + 16, gk + 8);
    const unsigned short* gv = gV + kb0;
    async_b128(lv[buf],      gv);
    async_b128(lv[buf] + 16, gv + 8);
  };
  issueKV(0, 0);
  wait_async4();          // Q complete (in-order), tile0 may still be in flight
  __syncthreads();

  // preload Q A-fragments (K=0..31 and K=32..63)
  V16 aq[2];
  {
    int r  = wave * 16 + lane16;
    int kb = laneHalf * 8;
    aq[0].h[0] = *reinterpret_cast<const v8bf*>(&sQ[r][kb]);
    aq[0].h[1] = *reinterpret_cast<const v8bf*>(&sQ[r][kb + 16]);
    aq[1].h[0] = *reinterpret_cast<const v8bf*>(&sQ[r][kb + 32]);
    aq[1].h[1] = *reinterpret_cast<const v8bf*>(&sQ[r][kb + 48]);
  }

  v8f zero = {0.f, 0.f, 0.f, 0.f, 0.f, 0.f, 0.f, 0.f};
  v8f o[4];
  for (int ni = 0; ni < 4; ++ni) o[ni] = zero;
  float mrow[8], lsum[8];
  for (int r = 0; r < 8; ++r) { mrow[r] = -1e30f; lsum[r] = 0.0f; }

  auto tile_compute = [&](int buf) {
    // S tile: 16 rows x 64 keys per wave
    v8f st[4];
    for (int ni = 0; ni < 4; ++ni) {
      v8f a = zero;
      for (int kk = 0; kk < 2; ++kk) {
        V16 bk;
        int key = ni * 16 + lane16;
        int kb  = laneHalf * 8 + kk * 32;
        bk.h[0] = *reinterpret_cast<const v8bf*>(&sK[buf][key][kb]);
        bk.h[1] = *reinterpret_cast<const v8bf*>(&sK[buf][key][kb + 16]);
        a = wmma_bf16(aq[kk].v, bk.v, a);
      }
      st[ni] = a;
    }
    // online softmax (per row; rows live across 16-lane halves)
    for (int r = 0; r < 8; ++r) {
      float mx = fmaxf(fmaxf(st[0][r], st[1][r]), fmaxf(st[2][r], st[3][r]));
      for (int off = 8; off; off >>= 1) mx = fmaxf(mx, __shfl_xor(mx, off, 32));
      float mn   = fmaxf(mrow[r], mx);
      float corr = __expf(mrow[r] - mn);
      mrow[r] = mn;
      float rs = 0.0f;
      for (int ni = 0; ni < 4; ++ni) {
        float p = __expf(st[ni][r] - mn);
        st[ni][r] = p;
        rs += p;
      }
      for (int off = 8; off; off >>= 1) rs += __shfl_xor(rs, off, 32);
      lsum[r] = lsum[r] * corr + rs;
      for (int ni = 0; ni < 4; ++ni) o[ni][r] *= corr;
    }
    // stage P (own 16-row strip only; DS ops are in-order per wave)
    int prow0 = wave * 16 + laneHalf * 8;
    for (int ni = 0; ni < 4; ++ni)
      for (int r = 0; r < 8; ++r)
        sP[prow0 + r][ni * 16 + lane16] = f2bf(st[ni][r]);
    // reload P as A-fragments, ctx += P x V
    V16 ap[2];
    {
      int pr = wave * 16 + lane16;
      int kb = laneHalf * 8;
      ap[0].h[0] = *reinterpret_cast<const v8bf*>(&sP[pr][kb]);
      ap[0].h[1] = *reinterpret_cast<const v8bf*>(&sP[pr][kb + 16]);
      ap[1].h[0] = *reinterpret_cast<const v8bf*>(&sP[pr][kb + 32]);
      ap[1].h[1] = *reinterpret_cast<const v8bf*>(&sP[pr][kb + 48]);
    }
    for (int ni = 0; ni < 4; ++ni) {
      for (int kk = 0; kk < 2; ++kk) {
        V16 bv;
        int dim = ni * 16 + lane16;
        int kb  = laneHalf * 8 + kk * 32;
        bv.h[0] = *reinterpret_cast<const v8bf*>(&sVt[buf][dim][kb]);
        bv.h[1] = *reinterpret_cast<const v8bf*>(&sVt[buf][dim][kb + 16]);
        o[ni] = wmma_bf16(ap[kk].v, bv.v, o[ni]);
      }
    }
  };

  const int NT = S_LEN / 64;   // 32 tiles
  for (int i = 0; i < NT - 1; ++i) {          // steady state: branch-free
    issueKV((i + 1) * 64, (i & 1) ^ 1);
    wait_async4();                            // tile i landed
    __syncthreads();
    tile_compute(i & 1);
    __syncthreads();                          // buf free for refill next iter
  }
  wait_async0();                              // peeled last tile
  __syncthreads();
  tile_compute((NT - 1) & 1);

  // finalize: divide by l, write ctx bf16 back to [S,B,D]
  for (int r = 0; r < 8; ++r) {
    float inv = 1.0f / lsum[r];
    int srow = q0 + wave * 16 + laneHalf * 8 + r;
    size_t grow = (size_t)srow * B_SZ + bb;
    for (int ni = 0; ni < 4; ++ni) {
      int col = colBase + ni * 16 + lane16;
      ctx[grow * D_MODEL + col] = f2bf(o[ni][r] * inv);
    }
  }
}

// ---------------- layernorm (one block per row) ----------------
__global__ __launch_bounds__(256) void k_layernorm(const float* __restrict__ in,
                                                   const float* __restrict__ g,
                                                   const float* __restrict__ be,
                                                   float* __restrict__ outF,
                                                   unsigned short* __restrict__ outB) {
  __shared__ float red[256];
  const int row = blockIdx.x;
  const float* x = in + (size_t)row * D_MODEL;
  float vals[4];
  float s = 0.0f;
  for (int j = 0; j < 4; ++j) { vals[j] = x[threadIdx.x + j * 256]; s += vals[j]; }
  red[threadIdx.x] = s; __syncthreads();
  for (int off = 128; off > 0; off >>= 1) {
    if (threadIdx.x < off) red[threadIdx.x] += red[threadIdx.x + off];
    __syncthreads();
  }
  float mean = red[0] / (float)D_MODEL;
  __syncthreads();
  float vs = 0.0f;
  for (int j = 0; j < 4; ++j) { float d = vals[j] - mean; vs += d * d; }
  red[threadIdx.x] = vs; __syncthreads();
  for (int off = 128; off > 0; off >>= 1) {
    if (threadIdx.x < off) red[threadIdx.x] += red[threadIdx.x + off];
    __syncthreads();
  }
  float rstd = rsqrtf(red[0] / (float)D_MODEL + 1e-5f);
  for (int j = 0; j < 4; ++j) {
    int c = threadIdx.x + j * 256;
    float y = (vals[j] - mean) * rstd * g[c] + be[c];
    size_t idx = (size_t)row * D_MODEL + c;
    if (outF) outF[idx] = y;
    if (outB) outB[idx] = f2bf(y);
  }
}

// ---------------- driver ----------------
extern "C" void kernel_launch(void* const* d_in, const int* in_sizes, int n_in,
                              void* d_out, int out_size, void* d_ws, size_t ws_size,
                              hipStream_t stream) {
  const float* src = (const float*)d_in[0];
  const float* Wq  = (const float*)d_in[1];
  const float* bq  = (const float*)d_in[2];
  const float* Wk  = (const float*)d_in[3];
  const float* bk  = (const float*)d_in[4];
  const float* Wv  = (const float*)d_in[5];
  const float* bv  = (const float*)d_in[6];
  const float* Wo  = (const float*)d_in[7];
  const float* bo  = (const float*)d_in[8];
  const float* g1  = (const float*)d_in[9];
  const float* be1 = (const float*)d_in[10];
  const float* W1  = (const float*)d_in[11];
  const float* bf1 = (const float*)d_in[12];
  const float* W2  = (const float*)d_in[13];
  const float* bf2 = (const float*)d_in[14];
  const float* g2  = (const float*)d_in[15];
  const float* be2 = (const float*)d_in[16];
  float* out = (float*)d_out;

  char* ws = (char*)d_ws;
  size_t off = 0;
  auto alloc = [&](size_t bytes) -> void* {
    void* p = ws + off;
    off += (bytes + 255) & ~(size_t)255;
    return p;
  };

  const size_t DD = (size_t)D_MODEL * D_MODEL;   // 1M
  const size_t DF = (size_t)D_MODEL * FF;        // 4M
  const size_t RD = (size_t)NROW * D_MODEL;      // 4M
  unsigned short* wqT  = (unsigned short*)alloc(DD * 2);
  unsigned short* wkT  = (unsigned short*)alloc(DD * 2);
  unsigned short* wvT  = (unsigned short*)alloc(DD * 2);
  unsigned short* woT  = (unsigned short*)alloc(DD * 2);
  unsigned short* w1T  = (unsigned short*)alloc(DF * 2);
  unsigned short* w2T  = (unsigned short*)alloc(DF * 2);
  unsigned short* Xb   = (unsigned short*)alloc(RD * 2);   // reused as FFN hidden
  unsigned short* Qbb  = (unsigned short*)alloc(RD * 2);
  unsigned short* Kbb  = (unsigned short*)alloc(RD * 2);
  unsigned short* Vbb  = (unsigned short*)alloc(RD * 2);
  unsigned short* ctxb = (unsigned short*)alloc(RD * 2);
  unsigned short* vtb  = (unsigned short*)alloc(RD * 2);   // V transposed per (b,h)
  float*          x1f  = (float*)alloc(RD * 4);            // reused as y1
  float*          xf   = (float*)alloc(RD * 4);
  unsigned short* xb   = (unsigned short*)alloc(RD * 2);
  unsigned short* hb   = Xb;        // 32MB contiguous: Xb+Qbb+Kbb+Vbb (dead by then)
  float*          y1f  = x1f;
  (void)ws_size; (void)in_sizes; (void)n_in; (void)out_size;

  dim3 blk(256);
  k_f32_to_bf16<<<(int)(RD / 256), blk, 0, stream>>>(src, Xb, (int)RD);
  k_transpose_to_bf16<<<(int)(DD / 256), blk, 0, stream>>>(Wq, wqT, D_MODEL, D_MODEL);
  k_transpose_to_bf16<<<(int)(DD / 256), blk, 0, stream>>>(Wk, wkT, D_MODEL, D_MODEL);
  k_transpose_to_bf16<<<(int)(DD / 256), blk, 0, stream>>>(Wv, wvT, D_MODEL, D_MODEL);
  k_transpose_to_bf16<<<(int)(DD / 256), blk, 0, stream>>>(Wo, woT, D_MODEL, D_MODEL);
  k_transpose_to_bf16<<<(int)(DF / 256), blk, 0, stream>>>(W1, w1T, D_MODEL, FF);
  k_transpose_to_bf16<<<(int)(DF / 256), blk, 0, stream>>>(W2, w2T, FF, D_MODEL);

  const float scaling = 0.125f;   // HD^-0.5
  dim3 gD(D_MODEL / 128, NROW / 128);   // (8, 32)
  k_gemm<<<gD, blk, 0, stream>>>(Xb, wqT, bq, scaling, 0, nullptr, nullptr, Qbb,
                                 NROW, D_MODEL, D_MODEL);
  k_gemm<<<gD, blk, 0, stream>>>(Xb, wkT, bk, 1.0f, 0, nullptr, nullptr, Kbb,
                                 NROW, D_MODEL, D_MODEL);
  k_gemm<<<gD, blk, 0, stream>>>(Xb, wvT, bv, 1.0f, 0, nullptr, nullptr, Vbb,
                                 NROW, D_MODEL, D_MODEL);
  k_transpose_v<<<(int)(RD / 256), blk, 0, stream>>>(Vbb, vtb);

  dim3 gA(B_SZ * NH, S_LEN / 128);      // (32, 16)
  k_attention<<<gA, blk, 0, stream>>>(Qbb, Kbb, vtb, ctxb);

  // attn_out projection + residual (f32)
  k_gemm<<<gD, blk, 0, stream>>>(ctxb, woT, bo, 1.0f, 0, src, x1f, nullptr,
                                 NROW, D_MODEL, D_MODEL);
  k_layernorm<<<NROW, blk, 0, stream>>>(x1f, g1, be1, xf, xb);

  dim3 gF(FF / 128, NROW / 128);        // (32, 32)
  k_gemm<<<gF, blk, 0, stream>>>(xb, w1T, bf1, 1.0f, 1, nullptr, nullptr, hb,
                                 NROW, FF, D_MODEL);
  k_gemm<<<gD, blk, 0, stream>>>(hb, w2T, bf2, 1.0f, 0, xf, y1f, nullptr,
                                 NROW, D_MODEL, FF);
  k_layernorm<<<NROW, blk, 0, stream>>>(y1f, g2, be2, out, nullptr);
}
